// SelfAttention_13460427506079
// MI455X (gfx1250) — matrix operand order repaired and checked by reading, not compiled, not run on hardware
//
#include <hip/hip_runtime.h>

typedef __attribute__((ext_vector_type(2)))  float        v2f;
typedef __attribute__((ext_vector_type(8)))  float        v8f;
typedef __attribute__((ext_vector_type(16))) _Float16     v16h;
typedef __attribute__((ext_vector_type(8)))  unsigned int v8u;
typedef __attribute__((ext_vector_type(4)))  unsigned int v4u;
typedef __attribute__((ext_vector_type(4)))  int          v4i;
typedef __attribute__((ext_vector_type(8)))  int          v8i;

#define B_  2
#define C_  64
#define P_  9216
#define D_  8
#define KT  32   // key tile
#define QW  16   // queries per wave
#define NW  4    // waves per block

// Rotate-butterfly max across a 16-lane DPP row (wave32: both halves reduce
// independently, matching the WMMA D-layout half-wave groups).  Pure VALU.
template <int CTRL>
__device__ __forceinline__ float fmax_dpp(float v) {
    int o = __builtin_amdgcn_update_dpp(0, __builtin_bit_cast(int, v),
                                        CTRL, 0xf, 0xf, true);
    return fmaxf(v, __builtin_bit_cast(float, o));
}
__device__ __forceinline__ float rowmax16(float v) {
    v = fmax_dpp<0x128>(v);   // ROW_ROR:8
    v = fmax_dpp<0x124>(v);   // ROW_ROR:4
    v = fmax_dpp<0x122>(v);   // ROW_ROR:2
    v = fmax_dpp<0x121>(v);   // ROW_ROR:1
    return v;
}

// ---------------------------------------------------------------------------
// Tensor Data Mover: 2D tile load (tile_d0 x tile_d1) from a 2D tensor with
// row stride stride0 (all sizes in data_size units).  D# per CDNA5 ISA:
//   group0: {count=1, lds_addr, global_addr[56:0], type=2}
//   group1: {data_size, tensor_dim0/1, tile_dim0/1/2, tensor_dim0/1_stride}
//   groups 2/3: zero (tensor is <= 2D)
// ---------------------------------------------------------------------------
__device__ __forceinline__ void tdm_load_2d(unsigned lds_addr, const void* gaddr,
                                            unsigned ds_code,     // 1=2B, 2=4B
                                            unsigned tensor_d0, unsigned tensor_d1,
                                            unsigned tile_d0, unsigned tile_d1,
                                            unsigned long long stride0)
{
    unsigned long long ga = (unsigned long long)gaddr;
    v4u g0;
    g0[0] = 1u;                                              // count=1
    g0[1] = lds_addr;                                        // LDS byte addr
    g0[2] = (unsigned)(ga & 0xFFFFFFFFu);                    // global[31:0]
    g0[3] = (unsigned)((ga >> 32) & 0x01FFFFFFu) | (2u << 30); // [56:32]|type=2
    v8i g1;
    g1[0] = (int)(ds_code << 16);                            // data_size
    g1[1] = (int)((tensor_d0 & 0xFFFFu) << 16);              // dim0[15:0]
    g1[2] = (int)((tensor_d0 >> 16) | ((tensor_d1 & 0xFFFFu) << 16));
    g1[3] = (int)((tensor_d1 >> 16) | (tile_d0 << 16));      // dim1 hi | tile0
    g1[4] = (int)tile_d1;                                    // tile1 | tile2=0
    g1[5] = (int)(stride0 & 0xFFFFFFFFu);                    // dim0_stride lo
    g1[6] = (int)((stride0 >> 32) & 0xFFFFu);                // hi | dim1_str=0
    g1[7] = 0;
    v4i z4 = {0, 0, 0, 0};
#if __clang_major__ >= 23
    v8i z8 = {0, 0, 0, 0, 0, 0, 0, 0};
    __builtin_amdgcn_tensor_load_to_lds(g0, g1, z4, z4, z8, 0);
#else
    __builtin_amdgcn_tensor_load_to_lds(g0, g1, z4, z4, 0);
#endif
}

// ---------------------------------------------------------------------------
// Stage 1: per-pixel 1x1-conv projections  q [B,P,8] f32, k [B,8,P] f32,
// v [B,64,P] f16.  ~190 MFLOP, pure VALU, trivially L2 resident.
// ---------------------------------------------------------------------------
__global__ __launch_bounds__(128) void qkv_kernel(
    const float* __restrict__ x,
    const float* __restrict__ Wq, const float* __restrict__ bq,
    const float* __restrict__ Wk, const float* __restrict__ bk,
    const float* __restrict__ Wv, const float* __restrict__ bv,
    float* __restrict__ qb, float* __restrict__ kb, _Float16* __restrict__ vb)
{
    int idx = blockIdx.x * 128 + threadIdx.x;      // 144 blocks -> B*P threads
    int b = idx / P_;
    int p = idx - b * P_;

    float xr[C_];
    #pragma unroll
    for (int c = 0; c < C_; ++c) xr[c] = x[(b * C_ + c) * P_ + p];

    #pragma unroll
    for (int d = 0; d < D_; ++d) {
        float aq = bq[d], ak = bk[d];
        #pragma unroll
        for (int c = 0; c < C_; ++c) {
            aq += Wq[d * C_ + c] * xr[c];
            ak += Wk[d * C_ + c] * xr[c];
        }
        qb[(b * P_ + p) * D_ + d] = aq;
        kb[(b * D_ + d) * P_ + p] = ak;
    }

    #pragma unroll 8
    for (int o = 0; o < C_; ++o) {
        float a = bv[o];
        #pragma unroll
        for (int c = 0; c < C_; ++c) a += Wv[o * C_ + c] * xr[c];
        vb[(b * C_ + o) * P_ + p] = (_Float16)a;
    }
}

// ---------------------------------------------------------------------------
// Stage 2: flash attention with double-buffered TDM pipeline.
// One wave = 16 query rows; block (4 waves) shares K/V key-tiles (32 keys)
// staged by the Tensor Data Mover into ping-pong LDS buffers; tile i+1
// streams in while tile i is computed (TDM ops complete in order per wave,
// so s_wait_tensorcnt 2 releases the older tile).
//   QK^T : v_wmma_f32_16x16x4_f32 (exact f32; 2 wmma per 16x16 logit tile)
//   P@V  : v_wmma_f32_16x16x32_f16, 4 channel groups of 16
//   l    : 5th f16 WMMA vs all-ones B (row sums in the softmax-state layout)
//   max  : DPP ROW_ROR butterfly (VALU only)
// ---------------------------------------------------------------------------
__global__ __launch_bounds__(128) void attn_kernel(
    const float* __restrict__ x,
    const float* __restrict__ qb, const float* __restrict__ kb,
    const _Float16* __restrict__ vb, const float* __restrict__ gamma,
    float* __restrict__ out)
{
    __shared__ __align__(128) float    kt[2][D_ * KT];       // 2x 8 x 32 f32
    __shared__ __align__(128) unsigned vt[2][C_ * KT / 2];   // 2x 64 x 32 f16
    __shared__ __align__(128) float    pbuf[NW][QW * KT];    // per-wave transp.

    const int tid  = threadIdx.x;
    const int lane = tid & 31;
    const int wid  = tid >> 5;
    const int blk  = blockIdx.x;
    const int bpb  = P_ / (QW * NW);          // 144 query-blocks per batch
    const int b    = blk / bpb;
    const int qbase = (blk - b * bpb) * (QW * NW) + wid * QW;

    const int ln16 = lane & 15;
    const int hi   = lane >> 4;               // half-wave select
    const int kd0  = hi ? 2 : 0;              // f32 WMMA K start for this half

    // Q A-fragments (f32 16x16x4 layout): lane = M row, VGPRs = K, halves K+2
    const float* qrow = qb + (size_t)(b * P_ + qbase + ln16) * D_;
    v2f aq0, aq1;
    aq0.x = qrow[kd0];     aq0.y = qrow[kd0 + 1];      // K = 0..3
    aq1.x = qrow[4 + kd0]; aq1.y = qrow[5 + kd0];      // K = 4..7

    v8f O[4];
    #pragma unroll
    for (int i = 0; i < 4; ++i) O[i] = (v8f)0.0f;
    v8f   Lacc = (v8f)0.0f;                    // row sums via ones-WMMA
    float mprev[8];
    #pragma unroll
    for (int r = 0; r < 8; ++r) mprev[r] = -3.0e38f;

    v16h ones;
    #pragma unroll
    for (int i = 0; i < 16; ++i) ones[i] = (_Float16)1.0f;

    const float*    kbb   = kb + (size_t)b * D_ * P_;
    const _Float16* vbb16 = vb + (size_t)b * C_ * P_;
    const unsigned  kt_lds0 = (unsigned)(unsigned long long)&kt[0][0];
    const unsigned  kt_lds1 = (unsigned)(unsigned long long)&kt[1][0];
    const unsigned  vt_lds0 = (unsigned)(unsigned long long)&vt[0][0];
    const unsigned  vt_lds1 = (unsigned)(unsigned long long)&vt[1][0];

    // prologue: tile 0 into buffer 0
    if (wid == 0) {
        tdm_load_2d(kt_lds0, kbb,   /*f32*/2, P_, D_, KT, D_, P_);
        tdm_load_2d(vt_lds0, vbb16, /*f16*/1, P_, C_, KT, C_, P_);
    }

    for (int it = 0; it < P_ / KT; ++it) {
        const int j0  = it * KT;
        const int cur = it & 1;
        __syncthreads();   // all waves finished reading buf[cur^1] (iter it-1)
        if (wid == 0) {    // one wave drives the TDM (EXEC-independent issue)
            if (j0 + KT < P_) {
                tdm_load_2d(cur ? kt_lds0 : kt_lds1, kbb + j0 + KT,
                            /*f32*/2, P_, D_, KT, D_, P_);
                tdm_load_2d(cur ? vt_lds0 : vt_lds1, vbb16 + j0 + KT,
                            /*f16*/1, P_, C_, KT, C_, P_);
                __builtin_amdgcn_s_wait_tensorcnt(2);  // tile `it` complete
            } else {
                __builtin_amdgcn_s_wait_tensorcnt(0);  // last tile complete
            }
        }
        __syncthreads();   // buf[cur] ready for all waves
        const float*    ktc = kt[cur];
        const unsigned* vtc = vt[cur];

        // ---- QK^T: two 16x16 f32 logit tiles (keys j0..+15, j0+16..+31) ----
        v8f e0 = (v8f)0.0f, e1 = (v8f)0.0f;
        {
            v2f bk0, bk1;
            bk0.x = ktc[(kd0)*KT + ln16];     bk0.y = ktc[(kd0 + 1) * KT + ln16];
            bk1.x = ktc[(4 + kd0)*KT + ln16]; bk1.y = ktc[(5 + kd0) * KT + ln16];
            e0 = __builtin_amdgcn_wmma_f32_16x16x4_f32(false, aq0, false, bk0,
                                                       (short)0, e0, false, false);
            e0 = __builtin_amdgcn_wmma_f32_16x16x4_f32(false, aq1, false, bk1,
                                                       (short)0, e0, false, false);
            bk0.x = ktc[(kd0)*KT + 16 + ln16];     bk0.y = ktc[(kd0 + 1) * KT + 16 + ln16];
            bk1.x = ktc[(4 + kd0)*KT + 16 + ln16]; bk1.y = ktc[(5 + kd0) * KT + 16 + ln16];
            e1 = __builtin_amdgcn_wmma_f32_16x16x4_f32(false, aq0, false, bk0,
                                                       (short)0, e1, false, false);
            e1 = __builtin_amdgcn_wmma_f32_16x16x4_f32(false, aq1, false, bk1,
                                                       (short)0, e1, false, false);
        }

        // ---- online softmax: row max via DPP butterfly, rescale O and L ----
        float p0[8], p1[8];
        #pragma unroll
        for (int r = 0; r < 8; ++r) {
            float t    = rowmax16(fmaxf(e0[r], e1[r]));
            float mnew = fmaxf(mprev[r], t);
            float al   = __expf(mprev[r] - mnew);
            p0[r] = __expf(e0[r] - mnew);
            p1[r] = __expf(e1[r] - mnew);
            mprev[r] = mnew;
            Lacc[r] *= al;
            #pragma unroll
            for (int cg = 0; cg < 4; ++cg) O[cg][r] *= al;
        }

        // ---- transpose P tile (D layout -> f16 A layout) via private LDS ----
        float* pb = pbuf[wid];
        #pragma unroll
        for (int r = 0; r < 8; ++r) {
            pb[(r + 8 * hi) * KT + ln16]      = p0[r];
            pb[(r + 8 * hi) * KT + 16 + ln16] = p1[r];
        }
        asm volatile("s_wait_dscnt 0" ::: "memory");   // wave-internal LDS sync

        v16h pa;
        {
            const v2f* prow = (const v2f*)(pb + ln16 * KT);
            const int  kb2  = hi ? 4 : 0;              // float2 units
            v8u pw;
            #pragma unroll
            for (int i = 0; i < 4; ++i) {
                v2f w0 = prow[kb2 + i];                // K = kb0+2i, kb0+2i+1
                v2f w1 = prow[kb2 + 8 + i];            // K = kb0+16+2i, +1
                pw[i]     = __builtin_bit_cast(unsigned,
                               __builtin_amdgcn_cvt_pkrtz(w0.x, w0.y));
                pw[4 + i] = __builtin_bit_cast(unsigned,
                               __builtin_amdgcn_cvt_pkrtz(w1.x, w1.y));
            }
            pa = __builtin_bit_cast(v16h, pw);
        }

        // ---- P @ V: 4 channel groups + ones-column row-sum WMMA ----
        #pragma unroll
        for (int cg = 0; cg < 4; ++cg) {
            v8u bw;
            const unsigned* vrow = vtc + (cg * 16 + ln16) * 16 + (hi ? 4 : 0);
            #pragma unroll
            for (int v = 0; v < 8; ++v) bw[v] = vrow[v + (v & 4)];
            v16h bvf = __builtin_bit_cast(v16h, bw);
            O[cg] = __builtin_amdgcn_wmma_f32_16x16x32_f16(false, pa, false, bvf,
                                                           (short)0, O[cg],
                                                           false, false);
        }
        Lacc = __builtin_amdgcn_wmma_f32_16x16x32_f16(false, pa, false, ones,
                                                      (short)0, Lacc,
                                                      false, false);
    }

    // ---- epilogue: normalize, gamma*out + x ----
    const float g = gamma[0];
    #pragma unroll
    for (int r = 0; r < 8; ++r) {
        float inv = 1.0f / Lacc[r];
        int p = qbase + r + 8 * hi;
        #pragma unroll
        for (int cg = 0; cg < 4; ++cg) {
            int c = cg * 16 + ln16;
            size_t idx = (size_t)(b * C_ + c) * P_ + p;
            out[idx] = g * (O[cg][r] * inv) + x[idx];
        }
    }
}

// ---------------------------------------------------------------------------
extern "C" void kernel_launch(void* const* d_in, const int* in_sizes, int n_in,
                              void* d_out, int out_size, void* d_ws, size_t ws_size,
                              hipStream_t stream) {
    const float* x     = (const float*)d_in[0];
    const float* Wq    = (const float*)d_in[1];
    const float* bq    = (const float*)d_in[2];
    const float* Wk    = (const float*)d_in[3];
    const float* bk    = (const float*)d_in[4];
    const float* Wv    = (const float*)d_in[5];
    const float* bv    = (const float*)d_in[6];
    const float* gamma = (const float*)d_in[7];
    float* out = (float*)d_out;

    char* ws = (char*)d_ws;
    float*    qb = (float*)ws;                                     // B*P*8 f32
    float*    kb = (float*)(ws + (size_t)B_ * P_ * D_ * 4);        // B*8*P f32
    _Float16* vb = (_Float16*)(ws + (size_t)2 * B_ * P_ * D_ * 4); // B*64*P f16

    qkv_kernel<<<(B_ * P_) / 128, 128, 0, stream>>>(x, Wq, bq, Wk, bk, Wv, bv,
                                                    qb, kb, vb);
    attn_kernel<<<B_ * (P_ / (QW * NW)), 128, 0, stream>>>(x, qb, kb, vb,
                                                           gamma, out);
}
